// BornSeries_21569325760983
// MI455X (gfx1250) — matrix operand order repaired and checked by reading, not compiled, not run on hardware
//
#include <hip/hip_runtime.h>
#include <math.h>

#define NN   4
#define HH   256
#define WWW  256
#define CC   8
#define CKK  2
#define WR   129               // W/2 + 1
#define NP   (NN*HH*WWW)       // 262144 pixels
#define ITERS 4

typedef __attribute__((ext_vector_type(16))) _Float16 v16h;
typedef __attribute__((ext_vector_type(8)))  float    v8f;
typedef unsigned int v4u  __attribute__((ext_vector_type(4)));
typedef int          v4i  __attribute__((ext_vector_type(4)));
typedef int          v8it __attribute__((ext_vector_type(8)));

// ---------------------------------------------------------------------------
// CDNA5 WMMA helper: D = A(16x32 f16) * B(32x16 f16) + C(16x16 f32)
// ---------------------------------------------------------------------------
__device__ __forceinline__ v8f wmma16(v16h a, v16h b, v8f c) {
  return __builtin_amdgcn_wmma_f32_16x16x32_f16(
      /*neg_a=*/false, a, /*neg_b=*/false, b,
      /*c_mod=*/(short)0, c, /*reuse_a=*/false, /*reuse_b=*/false);
}
__device__ __forceinline__ v8f zero8() { v8f z; for (int i = 0; i < 8; i++) z[i] = 0.f; return z; }

// Wave32 A-matrix (16x32 f16) lane/element -> (m, k) per ISA 7.12.2
__device__ __forceinline__ void a_coords(int lane, int e, int& m, int& k) {
  int g = (lane >> 4) & 1;
  m = lane & 15;
  k = ((e >> 3) << 4) + (g << 3) + (e & 7);
}
// Wave32 B-matrix (32x16 f16) lane/element -> (k, n)
__device__ __forceinline__ void b_coords(int lane, int e, int& k, int& n) {
  int g = (lane >> 4) & 1;
  n = lane & 15;
  k = ((e >> 3) << 4) + (g << 3) + (e & 7);
}

// Branch-free tanh-gelu: tanh(u) = 1 - 2/(1+exp(2u)); exp->v_exp_f32 (TRANS,
// co-executes with matrix ops), rcp->v_rcp_f32. No EXEC-mask divergence.
__device__ __forceinline__ float gelu_t(float x) {
  float u = 0.7978845608028654f * (x + 0.044715f * x * x * x);
  float e = __expf(2.f * u);
  float t = 1.f - 2.f * __builtin_amdgcn_rcpf(1.f + e);
  return 0.5f * x * (1.f + t);
}

// ---------------------------------------------------------------------------
// Tensor Data Mover: stage a 2D tile (rows x row_elems, element 2^dsize_code
// bytes, row stride stride_elems) between global and a compact LDS buffer.
// D# packing per CDNA5 ISA. Issued by one wave,
// tracked with TENSORcnt.
// ---------------------------------------------------------------------------
__device__ __forceinline__ unsigned lds_off(const void* p) {
  return (unsigned)(size_t)p;      // low 32 bits of LDS aperture addr = offset
}

__device__ __forceinline__ void tdm_copy(unsigned lds_addr, const void* gptr,
                                         unsigned row_elems, unsigned rows,
                                         unsigned long long stride_elems,
                                         unsigned dsize_code, bool is_store) {
  unsigned long long ga = (unsigned long long)gptr;
  v4u g0;
  g0.x = 1u;                                            // count=1, user D#
  g0.y = lds_addr;                                      // lds_addr [63:32]
  g0.z = (unsigned)(ga & 0xffffffffu);                  // global_addr lo
  g0.w = (unsigned)((ga >> 32) & 0x01ffffffu) | (2u << 30);  // ga[56:32] | type=2
  v8it g1;
  g1[0] = (int)(dsize_code << 16);                      // wg_mask=0, data_size
  g1[1] = (int)((row_elems & 0xffffu) << 16);           // tensor_dim0 lo16
  g1[2] = (int)((row_elems >> 16) & 0xffffu)            // tensor_dim0 hi16
        | (int)((rows & 0xffffu) << 16);                // tensor_dim1 lo16
  g1[3] = (int)((rows >> 16) & 0xffffu)                 // tensor_dim1 hi16
        | (int)((row_elems & 0xffffu) << 16);           // tile_dim0
  g1[4] = (int)(rows & 0xffffu);                        // tile_dim1 (tile_dim2=0)
  g1[5] = (int)(stride_elems & 0xffffffffu);            // tensor_dim0_stride lo32
  g1[6] = (int)((stride_elems >> 32) & 0xffffu);        // stride hi16, dim1_stride=0
  g1[7] = 0;
  v4i gz = {0, 0, 0, 0};
#if defined(__clang_major__) && (__clang_major__ >= 23)
  v8it gz8 = {0, 0, 0, 0, 0, 0, 0, 0};
  if (is_store) __builtin_amdgcn_tensor_store_from_lds(g0, g1, gz, gz, gz8, 0);
  else          __builtin_amdgcn_tensor_load_to_lds(g0, g1, gz, gz, gz8, 0);
#else
  if (is_store) __builtin_amdgcn_tensor_store_from_lds(g0, g1, gz, gz, 0);
  else          __builtin_amdgcn_tensor_load_to_lds(g0, g1, gz, gz, 0);
#endif
}

// ---------------------------------------------------------------------------
// Spectral filter MLP (scalar; only 256*129 bins)
// ---------------------------------------------------------------------------
__device__ void mlp_proj(float in0, float in1,
                         const float* __restrict__ w1, const float* __restrict__ b1,
                         const float* __restrict__ w2, const float* __restrict__ b2,
                         const float* __restrict__ w3, const float* __restrict__ b3,
                         float* __restrict__ outp) {        // writes outp[o*2]
  float h1[16], h2[16];
  for (int o = 0; o < 16; o++)
    h1[o] = gelu_t(in0 * w1[o] + in1 * w1[16 + o] + b1[o]);
  for (int o = 0; o < 16; o++) {
    float s = b2[o];
    for (int i = 0; i < 16; i++) s += h1[i] * w2[i * 16 + o];
    h2[o] = gelu_t(s);
  }
  for (int o = 0; o < 64; o++) {
    float s = b3[o];
    for (int i = 0; i < 16; i++) s += h2[i] * w3[i * 64 + o];
    outp[(size_t)o * 2] = s * (1.f / 64.f);
  }
}

__global__ void filt_gen(const float* fr_w1, const float* fr_b1, const float* fr_w2,
                         const float* fr_b2, const float* fr_w3, const float* fr_b3,
                         const float* fi_w1, const float* fi_b1, const float* fi_w2,
                         const float* fi_b2, const float* fi_w3, const float* fi_b3,
                         float* filt /* (256*129*64) float2 interleaved */) {
  int idx = blockIdx.x * blockDim.x + threadIdx.x;
  if (idx >= 256 * WR) return;
  int a = idx / WR, b = idx % WR;
  float in0 = (float)b;                                   // kx
  float in1 = (a < 128) ? (float)a : (float)(a - 256);    // ky
  float* base = filt + (size_t)idx * 128;
  mlp_proj(in0, in1, fr_w1, fr_b1, fr_w2, fr_b2, fr_w3, fr_b3, base + 0);
  mlp_proj(in0, in1, fi_w1, fi_b1, fi_w2, fi_b2, fi_w3, fi_b3, base + 1);
}

// ---------------------------------------------------------------------------
// Per-pixel MLP -> 8x8 matrix field, on WMMA. One wave handles 16 pixels.
// ---------------------------------------------------------------------------
__global__ void __launch_bounds__(32)
matgen(const float* __restrict__ kin,
       const float* __restrict__ w1, const float* __restrict__ b1,
       const float* __restrict__ w2, const float* __restrict__ b2,
       const float* __restrict__ w3, const float* __restrict__ b3,
       float* __restrict__ outm /* NP x 64 */) {
  __shared__ float sh[16 * 16];
  int lane = threadIdx.x;
  size_t px0 = (size_t)blockIdx.x * 16;

  v16h A, B;
  for (int e = 0; e < 16; e++) {
    int m, k; a_coords(lane, e, m, k);
    A[e] = (_Float16)((k < CKK) ? kin[(px0 + m) * CKK + k] : 0.f);
  }
  for (int e = 0; e < 16; e++) {
    int k, n; b_coords(lane, e, k, n);
    B[e] = (_Float16)((k < CKK) ? w1[k * 16 + n] : 0.f);
  }
  v8f c = zero8();
  c = wmma16(A, B, c);
  {
    int g = (lane >> 4) & 1, n = lane & 15;
    for (int i = 0; i < 8; i++) sh[(8 * g + i) * 16 + n] = gelu_t(c[i] + b1[n]);
  }
  __syncthreads();

  for (int e = 0; e < 16; e++) {
    int m, k; a_coords(lane, e, m, k);
    A[e] = (_Float16)((k < 16) ? sh[m * 16 + k] : 0.f);
  }
  for (int e = 0; e < 16; e++) {
    int k, n; b_coords(lane, e, k, n);
    B[e] = (_Float16)((k < 16) ? w2[k * 16 + n] : 0.f);
  }
  c = zero8();
  c = wmma16(A, B, c);
  __syncthreads();
  {
    int g = (lane >> 4) & 1, n = lane & 15;
    for (int i = 0; i < 8; i++) sh[(8 * g + i) * 16 + n] = gelu_t(c[i] + b2[n]);
  }
  __syncthreads();

  for (int e = 0; e < 16; e++) {
    int m, k; a_coords(lane, e, m, k);
    A[e] = (_Float16)((k < 16) ? sh[m * 16 + k] : 0.f);
  }
  int g = (lane >> 4) & 1, n = lane & 15;
  for (int t = 0; t < 4; t++) {
    for (int e = 0; e < 16; e++) {
      int k, nn; b_coords(lane, e, k, nn);
      B[e] = (_Float16)((k < 16) ? w3[k * 64 + 16 * t + nn] : 0.f);
    }
    v8f d = zero8();
    d = wmma16(A, B, d);
    for (int i = 0; i < 8; i++)
      outm[(px0 + 8 * g + i) * 64 + 16 * t + n] = (d[i] + b3[16 * t + n]) * (1.f / 64.f);
  }
}

// ---------------------------------------------------------------------------
// local = 2u - u*gmat ; du = u*dmat
// ---------------------------------------------------------------------------
__global__ void local_du(const float* __restrict__ u, const float* __restrict__ gmat,
                         const float* __restrict__ dmat,
                         float* __restrict__ lo, float* __restrict__ du) {
  size_t p = (size_t)blockIdx.x * blockDim.x + threadIdx.x;
  if (p >= NP) return;
  if (p + 4096 < NP) {
    __builtin_prefetch(gmat + (p + 4096) * 64, 0, 1);   // global_prefetch_b8
    __builtin_prefetch(dmat + (p + 4096) * 64, 0, 1);
  }
  float uv[8];
  for (int j = 0; j < 8; j++) uv[j] = u[p * 8 + j];
  const float* gm = gmat + p * 64;
  const float* dm = dmat + p * 64;
  for (int k = 0; k < 8; k++) {
    float sg = 0.f, sd = 0.f;
    for (int j = 0; j < 8; j++) { sg += uv[j] * gm[j * 8 + k]; sd += uv[j] * dm[j * 8 + k]; }
    lo[p * 8 + k] = 2.f * uv[k] - sg;
    du[p * 8 + k] = sd;
  }
}

// ---------------------------------------------------------------------------
// rfft along W as WMMA DFT-GEMM. TDM stages the contiguous (W,C) row in LDS.
// ---------------------------------------------------------------------------
__global__ void __launch_bounds__(96)
dft_w_fwd(const float* __restrict__ src /* N,H,W,C real */,
          float2* __restrict__ dst      /* N,H,129,C complex */) {
  __shared__ float2 tw[256];
  __shared__ float  shrow[WWW * CC];                     // 8 KB
  for (int t = threadIdx.x; t < 256; t += 96) {
    float a = 6.283185307179586f * (float)t / 256.f;
    tw[t] = make_float2(cosf(a), sinf(a));
  }
  size_t base = (size_t)blockIdx.x * (WWW * CC);
  if (threadIdx.x < 32) {
    tdm_copy(lds_off(shrow), src + base, WWW * CC, 1, WWW * CC, 2, false);
    __builtin_amdgcn_s_wait_tensorcnt(0);
  }
  __syncthreads();

  int wave = threadIdx.x >> 5, lane = threadIdx.x & 31;
  size_t ob = (size_t)blockIdx.x * (WR * CC);
  v8f aR[3], aI[3];
  for (int t = 0; t < 3; t++) { aR[t] = zero8(); aI[t] = zero8(); }
  for (int ks = 0; ks < 8; ks++) {
    v16h Br;
    for (int e = 0; e < 16; e++) {
      int k, n; b_coords(lane, e, k, n);
      int w_ = ks * 32 + k;
      Br[e] = (_Float16)((n < 8) ? shrow[w_ * 8 + n] : 0.f);
    }
    for (int t = 0; t < 3; t++) {
      int mt = wave * 3 + t;
      v16h Ar, Ai;                            // e^{-i*2pi*row*col/256}
      for (int e = 0; e < 16; e++) {
        int m, k; a_coords(lane, e, m, k);
        int row = mt * 16 + m, col = ks * 32 + k;
        float2 wv = tw[(row * col) & 255];
        Ar[e] = (_Float16)wv.x;
        Ai[e] = (_Float16)(-wv.y);
      }
      aR[t] = wmma16(Ar, Br, aR[t]);
      aI[t] = wmma16(Ai, Br, aI[t]);
    }
  }
  int g = (lane >> 4) & 1, n = lane & 15;
  if (n < 8)
    for (int t = 0; t < 3; t++) {
      int mt = wave * 3 + t;
      for (int i = 0; i < 8; i++) {
        int row = mt * 16 + 8 * g + i;
        if (row < WR) dst[ob + (size_t)row * 8 + n] = make_float2(aR[t][i], aI[t][i]);
      }
    }
}

// ---------------------------------------------------------------------------
// Complex DFT along H. For fixed (n,kw) the (h,c) panel is strided by
// 129*64B per row -> TDM-gather it into a compact LDS panel, compute on
// WMMA, TDM-scatter the result panel back. 4 waves x 4 M-tiles.
// ---------------------------------------------------------------------------
__global__ void __launch_bounds__(128)
dft_h(const float2* __restrict__ src, float2* __restrict__ dst, float sgn, float scale) {
  __shared__ float2 tw[256];
  __shared__ float2 shin[HH * CC];                       // 16 KB
  __shared__ float2 shout[HH * CC];                      // 16 KB
  for (int t = threadIdx.x; t < 256; t += 128) {
    float a = 6.283185307179586f * (float)t / 256.f;
    tw[t] = make_float2(cosf(a), sinf(a));
  }
  int n0 = blockIdx.x / WR, kw = blockIdx.x % WR;
  const float2* gsrc = src + (((size_t)n0 * HH) * WR + kw) * 8;
  float2*       gdst = dst + (((size_t)n0 * HH) * WR + kw) * 8;
  if (threadIdx.x < 32) {
    tdm_copy(lds_off(shin), gsrc, CC, HH, (size_t)WR * CC, 3, false);
    __builtin_amdgcn_s_wait_tensorcnt(0);
  }
  __syncthreads();

  int wave = threadIdx.x >> 5, lane = threadIdx.x & 31;
  v8f aR[4], aI[4];
  for (int t = 0; t < 4; t++) { aR[t] = zero8(); aI[t] = zero8(); }
  for (int ks = 0; ks < 8; ks++) {
    v16h Br, Bi;
    for (int e = 0; e < 16; e++) {
      int k, n; b_coords(lane, e, k, n);
      int h_ = ks * 32 + k;
      float2 v = make_float2(0.f, 0.f);
      if (n < 8) v = shin[h_ * 8 + n];
      Br[e] = (_Float16)v.x;
      Bi[e] = (_Float16)v.y;
    }
    for (int t = 0; t < 4; t++) {
      int mt = wave * 4 + t;
      v16h Ar, Ai, An;                        // A = cos + i*sgn*sin; An = -Im(A)
      for (int e = 0; e < 16; e++) {
        int m, k; a_coords(lane, e, m, k);
        int row = mt * 16 + m, col = ks * 32 + k;
        float2 wv = tw[(row * col) & 255];
        float ci = sgn * wv.y;
        Ar[e] = (_Float16)wv.x;
        Ai[e] = (_Float16)ci;
        An[e] = (_Float16)(-ci);
      }
      aR[t] = wmma16(Ar, Br, aR[t]);          // Dr += Ar*Br - Ai*Bi
      aR[t] = wmma16(An, Bi, aR[t]);
      aI[t] = wmma16(Ar, Bi, aI[t]);          // Di += Ar*Bi + Ai*Br
      aI[t] = wmma16(Ai, Br, aI[t]);
    }
  }
  int g = (lane >> 4) & 1, n = lane & 15;
  if (n < 8)
    for (int t = 0; t < 4; t++) {
      int mt = wave * 4 + t;
      for (int i = 0; i < 8; i++) {
        int row = mt * 16 + 8 * g + i;
        shout[row * 8 + n] = make_float2(aR[t][i] * scale, aI[t][i] * scale);
      }
    }
  __syncthreads();
  if (threadIdx.x < 32) {
    tdm_copy(lds_off(shout), gdst, CC, HH, (size_t)WR * CC, 3, true);
    __builtin_amdgcn_s_wait_tensorcnt(0);     // LDS must stay valid until DMA done
  }
}

// ---------------------------------------------------------------------------
// Per-frequency 8x8 complex matvec with the Green's filter
// ---------------------------------------------------------------------------
__global__ void filter_mv(const float2* __restrict__ src, const float2* __restrict__ filt,
                          float2* __restrict__ dst) {
  size_t idx = (size_t)blockIdx.x * blockDim.x + threadIdx.x;
  if (idx >= (size_t)NN * HH * WR) return;
  size_t rem = idx % ((size_t)HH * WR);
  const float2* x  = src + idx * 8;
  const float2* fm = filt + rem * 64;
  float2 xr[8];
  for (int j = 0; j < 8; j++) xr[j] = x[j];
  float2* o = dst + idx * 8;
  for (int k = 0; k < 8; k++) {
    float re = 0.f, im = 0.f;
    for (int j = 0; j < 8; j++) {
      float2 f = fm[j * 8 + k];
      re += xr[j].x * f.x - xr[j].y * f.y;
      im += xr[j].x * f.y + xr[j].y * f.x;
    }
    o[k] = make_float2(re, im);
  }
}

// ---------------------------------------------------------------------------
// irfft along W. TDM stages the contiguous 129*C complex row; hermitian
// weights and 1/256 folded into B.
// ---------------------------------------------------------------------------
__global__ void __launch_bounds__(128)
dft_w_inv(const float2* __restrict__ src /* N,H,129,C */, float* __restrict__ dst) {
  __shared__ float2 tw[256];
  __shared__ float2 shsp[WR * CC];                       // 8.25 KB
  for (int t = threadIdx.x; t < 256; t += 128) {
    float a = 6.283185307179586f * (float)t / 256.f;
    tw[t] = make_float2(cosf(a), sinf(a));
  }
  size_t ib = (size_t)blockIdx.x * (WR * CC);
  if (threadIdx.x < 32) {
    tdm_copy(lds_off(shsp), src + ib, WR * CC, 1, WR * CC, 3, false);
    __builtin_amdgcn_s_wait_tensorcnt(0);
  }
  __syncthreads();

  int wave = threadIdx.x >> 5, lane = threadIdx.x & 31;
  size_t ob = (size_t)blockIdx.x * (WWW * CC);
  v8f acc[4];
  for (int t = 0; t < 4; t++) acc[t] = zero8();
  for (int ks = 0; ks < 5; ks++) {
    v16h Br, Bi;
    for (int e = 0; e < 16; e++) {
      int k, n; b_coords(lane, e, k, n);
      int kw = ks * 32 + k;
      float2 v = make_float2(0.f, 0.f);
      if (kw < WR && n < 8) {
        v = shsp[kw * 8 + n];
        float wgt = ((kw == 0 || kw == 128) ? 1.f : 2.f) * (1.f / 256.f);
        v.x *= wgt; v.y *= wgt;
      }
      Br[e] = (_Float16)v.x;
      Bi[e] = (_Float16)v.y;
    }
    for (int t = 0; t < 4; t++) {
      int mt = wave * 4 + t;
      v16h Ac, An;                            // Re(X e^{+i th}) = Xr cos - Xi sin
      for (int e = 0; e < 16; e++) {
        int m, k; a_coords(lane, e, m, k);
        int row = mt * 16 + m, col = ks * 32 + k;
        float2 wv = tw[(row * col) & 255];
        Ac[e] = (_Float16)wv.x;
        An[e] = (_Float16)(-wv.y);
      }
      acc[t] = wmma16(Ac, Br, acc[t]);
      acc[t] = wmma16(An, Bi, acc[t]);
    }
  }
  int g = (lane >> 4) & 1, n = lane & 15;
  if (n < 8)
    for (int t = 0; t < 4; t++) {
      int mt = wave * 4 + t;
      for (int i = 0; i < 8; i++)
        dst[ob + (size_t)(mt * 16 + 8 * g + i) * 8 + n] = acc[t][i];
    }
}

// ---------------------------------------------------------------------------
// u = local + gu * gmat ; final gelu(u + bias)
// ---------------------------------------------------------------------------
__global__ void combine(const float* __restrict__ lo, const float* __restrict__ gu,
                        const float* __restrict__ gmat, float* __restrict__ uo) {
  size_t p = (size_t)blockIdx.x * blockDim.x + threadIdx.x;
  if (p >= NP) return;
  if (p + 4096 < NP) __builtin_prefetch(gmat + (p + 4096) * 64, 0, 1);
  float gv[8];
  for (int j = 0; j < 8; j++) gv[j] = gu[p * 8 + j];
  const float* gm = gmat + p * 64;
  for (int k = 0; k < 8; k++) {
    float s = 0.f;
    for (int j = 0; j < 8; j++) s += gv[j] * gm[j * 8 + k];
    uo[p * 8 + k] = lo[p * 8 + k] + s;
  }
}

__global__ void out_gelu(const float* __restrict__ u, const float* __restrict__ bias,
                         float* __restrict__ out) {
  size_t i = (size_t)blockIdx.x * blockDim.x + threadIdx.x;
  if (i >= (size_t)NP * 8) return;
  out[i] = gelu_t(u[i] + bias[i & 7]);
}

// ---------------------------------------------------------------------------
extern "C" void kernel_launch(void* const* d_in, const int* in_sizes, int n_in,
                              void* d_out, int out_size, void* d_ws, size_t ws_size,
                              hipStream_t stream) {
  (void)in_sizes; (void)n_in; (void)out_size; (void)ws_size;
  const float* u_in = (const float*)d_in[0];
  const float* k_in = (const float*)d_in[1];
  const float* gw[6];  for (int i = 0; i < 6; i++) gw[i]  = (const float*)d_in[2 + i];
  const float* dw[6];  for (int i = 0; i < 6; i++) dw[i]  = (const float*)d_in[8 + i];
  const float* frw[6]; for (int i = 0; i < 6; i++) frw[i] = (const float*)d_in[14 + i];
  const float* fiw[6]; for (int i = 0; i < 6; i++) fiw[i] = (const float*)d_in[20 + i];
  const float* bias = (const float*)d_in[26];
  float* out = (float*)d_out;

  // ---- workspace carve (floats) ----
  float* ws = (float*)d_ws;
  size_t off = 0;
  float*  gmat  = ws + off; off += (size_t)NP * 64;
  float*  dmat  = ws + off; off += (size_t)NP * 64;
  float*  filtf = ws + off; off += (size_t)256 * WR * 64 * 2;
  float*  lo    = ws + off; off += (size_t)NP * 8;
  float*  du    = ws + off; off += (size_t)NP * 8;
  float*  ub    = ws + off; off += (size_t)NP * 8;
  float*  gu    = ws + off; off += (size_t)NP * 8;
  float2* s1    = (float2*)(ws + off); off += (size_t)NN * HH * WR * CC * 2;
  float2* s2    = (float2*)(ws + off); off += (size_t)NN * HH * WR * CC * 2;
  float2* filt  = (float2*)filtf;

  // ---- precompute: spectral filter + per-pixel gamma/delta matrices ----
  filt_gen<<<(256 * WR + 127) / 128, 128, 0, stream>>>(
      frw[0], frw[1], frw[2], frw[3], frw[4], frw[5],
      fiw[0], fiw[1], fiw[2], fiw[3], fiw[4], fiw[5], filtf);
  matgen<<<NP / 16, 32, 0, stream>>>(k_in, gw[0], gw[1], gw[2], gw[3], gw[4], gw[5], gmat);
  matgen<<<NP / 16, 32, 0, stream>>>(k_in, dw[0], dw[1], dw[2], dw[3], dw[4], dw[5], dmat);

  // ---- Born iterations ----
  for (int it = 0; it < ITERS; it++) {
    const float* ucur = (it == 0) ? u_in : ub;
    local_du<<<NP / 256, 256, 0, stream>>>(ucur, gmat, dmat, lo, du);
    dft_w_fwd<<<NN * HH, 96, 0, stream>>>(du, s1);
    dft_h<<<NN * WR, 128, 0, stream>>>(s1, s2, -1.f, 1.f);
    filter_mv<<<(NN * HH * WR + 127) / 128, 128, 0, stream>>>(s2, filt, s1);
    dft_h<<<NN * WR, 128, 0, stream>>>(s1, s2, +1.f, 1.f / 256.f);
    dft_w_inv<<<NN * HH, 128, 0, stream>>>(s2, gu);
    combine<<<NP / 256, 256, 0, stream>>>(lo, gu, gmat, ub);
  }
  out_gelu<<<(NP * 8 + 255) / 256, 256, 0, stream>>>(ub, bias, out);
}
